// RelPosMultiHeadSelfAttention_884763263170
// MI455X (gfx1250) — matrix operand order repaired and checked by reading, not compiled, not run on hardware
//
#include <hip/hip_runtime.h>

typedef __attribute__((ext_vector_type(16))) __bf16 v16bf;
typedef __attribute__((ext_vector_type(8)))  float  v8f;

#define T_SEQ   1024
#define DMODEL  512
#define NHEAD   8
#define DHEAD   64
#define RROWS   2047      // 2T-1 valid relative rows
#define RPAD    2048      // padded to WMMA tile multiple
#define BATCH   4
#define MTOK    (BATCH * T_SEQ)   // 4096
#define INV_SQRT_DH 0.125f

// ---------------- WMMA helpers (layouts per CDNA5 ISA §7.12.2) ---

__device__ __forceinline__ v8f wmma_bf16(v16bf a, v16bf b, v8f c) {
  // D = A(16x32 bf16) * B(32x16 bf16) + C(16x16 f32)
  return __builtin_amdgcn_wmma_f32_16x16x32_bf16(
      false, a, false, b, (short)0, c, false, false);
}

// A fragment 16x32 (MxK): lane<16 holds row M=lane, K = {0..7,16..23};
// lane>=16 holds row M=lane-16, K = {8..15,24..31}.  fp32 source + convert.
template <bool GUARD>
__device__ __forceinline__ v16bf frag_a_f32(const float* base, int ld, int row_lo,
                                            int k0, int lane, int mvalid) {
  const int half = lane >> 4;
  const int r = row_lo + (lane & 15);
  v16bf f;
  if (!GUARD || r < mvalid) {
    const float* p = base + (size_t)r * ld + k0 + half * 8;
#pragma unroll
    for (int i = 0; i < 8; ++i) f[i] = (__bf16)p[i];
#pragma unroll
    for (int i = 0; i < 8; ++i) f[8 + i] = (__bf16)p[16 + i];
  } else {
#pragma unroll
    for (int i = 0; i < 16; ++i) f[i] = (__bf16)0.0f;
  }
  return f;
}

// Same A fragment from a bf16 row-major source.
__device__ __forceinline__ v16bf frag_a_bf16(const __bf16* base, int ld, int row_lo,
                                             int k0, int lane) {
  const int half = lane >> 4;
  const __bf16* p = base + (size_t)(row_lo + (lane & 15)) * ld + k0 + half * 8;
  v16bf f;
#pragma unroll
  for (int i = 0; i < 8; ++i) f[i] = p[i];
#pragma unroll
  for (int i = 0; i < 8; ++i) f[8 + i] = p[16 + i];
  return f;
}

// A fragment from fp32 data in LDS (probabilities), row_lo == 0.
__device__ __forceinline__ v16bf frag_a_lds32(const float* base, int ld,
                                              int k0, int lane) {
  const int half = lane >> 4;
  const float* p = base + (lane & 15) * ld + k0 + half * 8;
  v16bf f;
#pragma unroll
  for (int i = 0; i < 8; ++i) f[i] = (__bf16)p[i];
#pragma unroll
  for (int i = 0; i < 8; ++i) f[8 + i] = (__bf16)p[16 + i];
  return f;
}

// B fragment 32x16 (KxN) for an NT product: memory holds B-rows (n, k) row-major,
// so lane<16 loads K = {0..7,16..23} of row n=lane; lane>=16 loads
// K = {8..15,24..31} of row n=lane-16.
__device__ __forceinline__ v16bf frag_b_bf16(const __bf16* base, int ld, int n_lo,
                                             int k0, int lane) {
  const __bf16* p = base + (size_t)(n_lo + (lane & 15)) * ld + k0 + (lane >> 4) * 8;
  v16bf f;
#pragma unroll
  for (int i = 0; i < 8; ++i) f[i] = p[i];
#pragma unroll
  for (int i = 0; i < 8; ++i) f[8 + i] = p[16 + i];
  return f;
}

__device__ __forceinline__ v16bf frag_b_f32(const float* base, int ld, int n_lo,
                                            int k0, int lane) {
  const float* p = base + (size_t)(n_lo + (lane & 15)) * ld + k0 + (lane >> 4) * 8;
  v16bf f;
#pragma unroll
  for (int i = 0; i < 8; ++i) f[i] = (__bf16)p[i];
#pragma unroll
  for (int i = 0; i < 8; ++i) f[8 + i] = (__bf16)p[16 + i];
  return f;
}

// ---------------- positional encodings ------------------------------------
// pe[r, 2j]   = sin((1023 - r) * 10000^(-2j/512))
// pe[r, 2j+1] = cos( ... ), rows >= 2047 zero-padded.
__global__ void pe_kernel(float* __restrict__ pe) {
  int idx = blockIdx.x * blockDim.x + threadIdx.x;
  if (idx >= RPAD * DMODEL) return;
  int r = idx / DMODEL;
  int d = idx - r * DMODEL;
  float v = 0.0f;
  if (r < RROWS) {
    int j = d >> 1;
    float inv_freq = __expf((float)j * (-2.0f / (float)DMODEL) * 9.210340371976184f);
    float ang = (1023.0f - (float)r) * inv_freq;
    v = (d & 1) ? cosf(ang) : sinf(ang);
  }
  pe[idx] = v;
}

// ---------------- generic NT GEMM: C[m,n] = sum_k A[m,k]*W[n,k] + bias[n] ---
// 128 threads = 4 waves; block tile 64x64; wave tile 32x32 (2x2 WMMA frags).
// MODE (compile-time): 0 = single bf16 out, 1 = dual bf16 out (+ubias/+vbias),
//                      2 = bf16 transposed (B,T,D)->(B,D,T), 3 = fp32 out.
#define GM_BF16   0
#define GM_DUAL   1
#define GM_TRANS  2
#define GM_F32    3

template <int MODE, bool GUARD>
__global__ void __launch_bounds__(128) gemm_nt(
    const float* __restrict__ A, const float* __restrict__ W,
    const float* __restrict__ bias,
    const float* __restrict__ ubias, const float* __restrict__ vbias,
    __bf16* __restrict__ out0, __bf16* __restrict__ out1,
    float* __restrict__ outf,
    int Mvalid) {
  const int wave = threadIdx.x >> 5;
  const int lane = threadIdx.x & 31;
  const int m0 = blockIdx.x * 64 + (wave >> 1) * 32;
  const int n0 = blockIdx.y * 64 + (wave & 1) * 32;

  v8f acc00 = {0.f,0.f,0.f,0.f,0.f,0.f,0.f,0.f};
  v8f acc01 = acc00, acc10 = acc00, acc11 = acc00;

#pragma unroll 2
  for (int k0 = 0; k0 < DMODEL; k0 += 32) {
    v16bf a0 = frag_a_f32<GUARD>(A, DMODEL, m0,      k0, lane, Mvalid);
    v16bf a1 = frag_a_f32<GUARD>(A, DMODEL, m0 + 16, k0, lane, Mvalid);
    v16bf b0 = frag_b_f32(W, DMODEL, n0,      k0, lane);
    v16bf b1 = frag_b_f32(W, DMODEL, n0 + 16, k0, lane);
    acc00 = wmma_bf16(a0, b0, acc00);
    acc01 = wmma_bf16(a0, b1, acc01);
    acc10 = wmma_bf16(a1, b0, acc10);
    acc11 = wmma_bf16(a1, b1, acc11);
  }

  const int half = lane >> 4;
  const int col  = lane & 15;
  v8f* accs[4] = {&acc00, &acc01, &acc10, &acc11};
#pragma unroll
  for (int t = 0; t < 4; ++t) {
    const int ti = t >> 1, tj = t & 1;
    const int n = n0 + tj * 16 + col;
    const float bn = bias[n];
#pragma unroll
    for (int j = 0; j < 8; ++j) {
      const int m = m0 + ti * 16 + j + 8 * half;
      if (GUARD && m >= Mvalid) continue;
      const float v = (*accs[t])[j] + bn;
      if (MODE == GM_F32) {
        outf[(size_t)m * DMODEL + n] = v;
      } else if (MODE == GM_DUAL) {
        out0[(size_t)m * DMODEL + n] = (__bf16)(v + ubias[n]);
        out1[(size_t)m * DMODEL + n] = (__bf16)(v + vbias[n]);
      } else if (MODE == GM_TRANS) {
        // store as (B, D, T): batch = m>>10, t = m&1023
        out0[((size_t)(m >> 10) * DMODEL + n) * T_SEQ + (m & (T_SEQ - 1))] = (__bf16)v;
      } else {
        out0[(size_t)m * DMODEL + n] = (__bf16)v;
      }
    }
  }
}

// ---------------- fused rel-pos attention ----------------------------------
// grid.x = B*H*(T/64); 128 threads = 4 waves; wave owns 16 q-rows.
// LDS: per-wave 16x1024 f32 score rows (256KB) + per-wave 16x32 rel scratch (8KB).
__global__ void __launch_bounds__(128) attn_kernel(
    const __bf16* __restrict__ Qu, const __bf16* __restrict__ Qv,
    const __bf16* __restrict__ Kb, const __bf16* __restrict__ Vt,
    const __bf16* __restrict__ Eb, float* __restrict__ Out) {
  extern __shared__ char smem_raw[];
  const int wave = threadIdx.x >> 5;
  const int lane = threadIdx.x & 31;
  float* scores = (float*)smem_raw + (size_t)wave * 16 * T_SEQ;
  float* relbuf = (float*)smem_raw + 4 * 16 * T_SEQ + wave * 16 * 32;

  const int tiles = T_SEQ / 64;
  const int bh = blockIdx.x / tiles;
  const int bb = bh / NHEAD;
  const int h  = bh - bb * NHEAD;
  const int q0 = (blockIdx.x - bh * tiles) * 64 + wave * 16;

  const __bf16* Qu_base = Qu + ((size_t)bb * T_SEQ) * DMODEL + h * DHEAD;
  const __bf16* Qv_base = Qv + ((size_t)bb * T_SEQ) * DMODEL + h * DHEAD;
  const __bf16* K_base  = Kb + ((size_t)bb * T_SEQ) * DMODEL + h * DHEAD;
  const __bf16* Vt_base = Vt + ((size_t)bb * DMODEL + h * DHEAD) * T_SEQ;
  const __bf16* E_base  = Eb + h * DHEAD;

  // Preload the wave's Q fragments (16 rows, dh=64 split into two K=32 chunks).
  v16bf aQu[2], aQv[2];
  aQu[0] = frag_a_bf16(Qu_base, DMODEL, q0, 0,  lane);
  aQu[1] = frag_a_bf16(Qu_base, DMODEL, q0, 32, lane);
  aQv[0] = frag_a_bf16(Qv_base, DMODEL, q0, 0,  lane);
  aQv[1] = frag_a_bf16(Qv_base, DMODEL, q0, 32, lane);

  const int half = lane >> 4;
  const int col  = lane & 15;

  // ---- pass 1: scores = (Qu.K^T + shift(Qv.E^T)) / sqrt(dh) ----
  for (int k0 = 0; k0 < T_SEQ; k0 += 16) {
    v8f sK  = {0.f,0.f,0.f,0.f,0.f,0.f,0.f,0.f};
    v8f sE0 = sK, sE1 = sK;
    const int r0 = k0 - q0 + (T_SEQ - 1) - 15;  // in [0, 2016]

    // Prefetch next tile's B-fragment rows (global_prefetch_b8, speculative).
    if (k0 + 16 < T_SEQ) {
      __builtin_prefetch(K_base + (size_t)(k0 + 16 + col) * DMODEL + (half ? 16 : 0), 0, 3);
      __builtin_prefetch(E_base + (size_t)(r0 + 32 + col) * DMODEL + (half ? 16 : 0), 0, 3);
    }
#pragma unroll
    for (int ks = 0; ks < 2; ++ks) {
      v16bf bK  = frag_b_bf16(K_base, DMODEL, k0,      ks * 32, lane);
      sK  = wmma_bf16(aQu[ks], bK,  sK);
      v16bf bE0 = frag_b_bf16(E_base, DMODEL, r0,      ks * 32, lane);
      sE0 = wmma_bf16(aQv[ks], bE0, sE0);
      v16bf bE1 = frag_b_bf16(E_base, DMODEL, r0 + 16, ks * 32, lane);
      sE1 = wmma_bf16(aQv[ks], bE1, sE1);
    }
    // Stash the 16x32 relative band in wave-private LDS...
#pragma unroll
    for (int j = 0; j < 8; ++j) {
      const int row = j + 8 * half;
      relbuf[row * 32 + col]      = sE0[j];
      relbuf[row * 32 + 16 + col] = sE1[j];
    }
    // ...then apply rel_to_abs: abs[q,k] = rel[q, k - q + T - 1] -> band col kk-qq+15.
#pragma unroll
    for (int j = 0; j < 8; ++j) {
      const int row = j + 8 * half;
      const float e = relbuf[row * 32 + (col - row + 15)];
      scores[row * T_SEQ + k0 + col] = (sK[j] + e) * INV_SQRT_DH;
    }
  }

  // ---- pass 2: row softmax in fp32 (wave-private rows, shfl reductions) ----
  for (int row = 0; row < 16; ++row) {
    float* sr = scores + row * T_SEQ;
    float x[32];
    float m = -3.402823466e38f;
#pragma unroll
    for (int i = 0; i < 32; ++i) { x[i] = sr[i * 32 + lane]; m = fmaxf(m, x[i]); }
#pragma unroll
    for (int off = 16; off > 0; off >>= 1) m = fmaxf(m, __shfl_xor(m, off, 32));
    float s = 0.0f;
#pragma unroll
    for (int i = 0; i < 32; ++i) { x[i] = __expf(x[i] - m); s += x[i]; }
#pragma unroll
    for (int off = 16; off > 0; off >>= 1) s += __shfl_xor(s, off, 32);
    const float inv = 1.0f / s;
#pragma unroll
    for (int i = 0; i < 32; ++i) sr[i * 32 + lane] = x[i] * inv;
  }

  // ---- pass 3: out = P @ V (V pre-transposed to (B,D,T) for contiguous frags) ----
  v8f acc[4];
#pragma unroll
  for (int n = 0; n < 4; ++n) acc[n] = (v8f){0.f,0.f,0.f,0.f,0.f,0.f,0.f,0.f};
  for (int ks = 0; ks < T_SEQ; ks += 32) {
    v16bf aP = frag_a_lds32(scores, T_SEQ, ks, lane);
#pragma unroll
    for (int n = 0; n < 4; ++n) {
      v16bf bV = frag_b_bf16(Vt_base, T_SEQ, n * 16, ks, lane);
      acc[n] = wmma_bf16(aP, bV, acc[n]);
    }
  }
#pragma unroll
  for (int n = 0; n < 4; ++n)
#pragma unroll
    for (int j = 0; j < 8; ++j) {
      const int row = j + 8 * half;
      Out[((size_t)bb * T_SEQ + q0 + row) * DMODEL + h * DHEAD + n * 16 + col] = acc[n][j];
    }
}

// ---------------- host-side orchestration ----------------------------------
extern "C" void kernel_launch(void* const* d_in, const int* in_sizes, int n_in,
                              void* d_out, int out_size, void* d_ws, size_t ws_size,
                              hipStream_t stream) {
  const float* query  = (const float*)d_in[0];
  const float* key    = (const float*)d_in[1];
  const float* value  = (const float*)d_in[2];
  const float* Wq     = (const float*)d_in[3];
  const float* bq     = (const float*)d_in[4];
  const float* Wk     = (const float*)d_in[5];
  const float* bk     = (const float*)d_in[6];
  const float* Wv     = (const float*)d_in[7];
  const float* bv     = (const float*)d_in[8];
  const float* Wp     = (const float*)d_in[9];
  const float* bp     = (const float*)d_in[10];
  const float* Wo     = (const float*)d_in[11];
  const float* bo     = (const float*)d_in[12];
  const float* u_bias = (const float*)d_in[13];
  const float* v_bias = (const float*)d_in[14];

  char* ws = (char*)d_ws;
  float*  pe   = (float*)ws;   ws += (size_t)RPAD * DMODEL * sizeof(float);
  __bf16* Qu   = (__bf16*)ws;  ws += (size_t)MTOK * DMODEL * sizeof(__bf16);
  __bf16* Qv   = (__bf16*)ws;  ws += (size_t)MTOK * DMODEL * sizeof(__bf16);
  __bf16* Kb   = (__bf16*)ws;  ws += (size_t)MTOK * DMODEL * sizeof(__bf16);
  __bf16* Vt   = (__bf16*)ws;  ws += (size_t)MTOK * DMODEL * sizeof(__bf16);
  __bf16* Eb   = (__bf16*)ws;  ws += (size_t)RPAD * DMODEL * sizeof(__bf16);
  float*  attn = (float*)ws;   ws += (size_t)MTOK * DMODEL * sizeof(float);

  const dim3 blk(128);
  const dim3 grid_full(MTOK / 64, DMODEL / 64);

  // 1. sinusoidal relative PE (fp32, zero-padded to 2048 rows)
  pe_kernel<<<(RPAD * DMODEL + 255) / 256, 256, 0, stream>>>(pe);

  // 2. projections (bf16 WMMA GEMMs, fp32 accumulate)
  gemm_nt<GM_DUAL, false><<<grid_full, blk, 0, stream>>>(
      query, Wq, bq, u_bias, v_bias, Qu, Qv, nullptr, MTOK);
  gemm_nt<GM_BF16, false><<<grid_full, blk, 0, stream>>>(
      key, Wk, bk, nullptr, nullptr, Kb, nullptr, nullptr, MTOK);
  gemm_nt<GM_TRANS, false><<<grid_full, blk, 0, stream>>>(
      value, Wv, bv, nullptr, nullptr, Vt, nullptr, nullptr, MTOK);
  gemm_nt<GM_BF16, true><<<dim3(RPAD / 64, DMODEL / 64), blk, 0, stream>>>(
      pe, Wp, bp, nullptr, nullptr, Eb, nullptr, nullptr, RROWS);

  // 3. fused rel-pos attention (264 KB dynamic LDS per workgroup)
  const size_t lds_bytes = (size_t)4 * 16 * T_SEQ * sizeof(float)
                         + (size_t)4 * 16 * 32 * sizeof(float);
  attn_kernel<<<dim3(BATCH * NHEAD * (T_SEQ / 64)), blk, lds_bytes, stream>>>(
      Qu, Qv, Kb, Vt, Eb, attn);

  // 4. output projection -> d_out (fp32)
  gemm_nt<GM_F32, false><<<grid_full, blk, 0, stream>>>(
      attn, Wo, bo, nullptr, nullptr, nullptr, nullptr, (float*)d_out, MTOK);
}
